// AttDecoder_36335423324434
// MI455X (gfx1250) — matrix operand order repaired and checked by reading, not compiled, not run on hardware
//
#include <hip/hip_runtime.h>
#include <hip/hip_bf16.h>
#include <math.h>

#define B_    8
#define C_    684
#define H_    16
#define W_    64
#define HW_   1024
#define M_    8192        // B*HW
#define HID_  256
#define AD_   512
#define V_    111
#define T_    36
#define KP_   704         // C padded to multiple of 32
#define KC_   128         // 11*11=121 padded to 128
#define LDK2  136         // LDS row pitch for K2 (breaks bank conflicts)

typedef __attribute__((ext_vector_type(16))) __bf16 v16bf;
typedef __attribute__((ext_vector_type(8)))  __bf16 v8bf;
typedef __attribute__((ext_vector_type(8)))  float  v8f;

__device__ __forceinline__ v16bf cat8(v8bf lo, v8bf hi) {
  return __builtin_shufflevector(lo, hi, 0,1,2,3,4,5,6,7,8,9,10,11,12,13,14,15);
}

// A fragment: 16x32 bf16 tile at (mrow, kb) from row-major A (pitch ld).
// Lanes 0-15: row M=lane, K = {kb..kb+15}
// Lanes16-31: row M=lane-16, K = {kb+16..kb+31}
__device__ __forceinline__ v16bf load_a(const __bf16* A, int ld, int mrow, int kb, int lane) {
  int r = lane & 15, hf = lane >> 4;
  const __bf16* p = A + (size_t)(mrow + r) * ld + kb + hf * 16;
  v8bf lo = *(const v8bf*)p;
  v8bf hi = *(const v8bf*)(p + 8);
  return cat8(lo, hi);
}

// B fragment: 32x16 bf16 tile (K x N) loaded from B^T stored (N x K) row-major.
// Lane holds column N = nb + lane%16, K range {kb..kb+15} (lanes 0-15) or {kb+16..kb+31}.
__device__ __forceinline__ v16bf load_bt(const __bf16* BT, int ld, int nb, int kb, int lane) {
  int n = nb + (lane & 15), hf = lane >> 4;
  const __bf16* p = BT + (size_t)n * ld + kb + hf * 16;
  v8bf lo = *(const v8bf*)p;
  v8bf hi = *(const v8bf*)(p + 8);
  return cat8(lo, hi);
}

// ---------------- one-time precompute ----------------

// per-batch: mask, avg-pool, hidden0=tanh(avg@initW.T+b), counting_ctx, zero alpha_sum
__global__ void k_init(const float* __restrict__ cnn, const float* __restrict__ cpred,
                       const float* __restrict__ imask,
                       const float* __restrict__ initW, const float* __restrict__ initb,
                       const float* __restrict__ cntW, const float* __restrict__ cntb,
                       float* __restrict__ mask_hw, float* __restrict__ hidden,
                       float* __restrict__ cctx, float* __restrict__ alpha_sum) {
  int b = blockIdx.x, tid = threadIdx.x;
  __shared__ float sm[HW_];
  __shared__ float avg[C_];
  __shared__ float red[256];
  for (int p = tid; p < HW_; p += 256) {
    int h = p >> 6, w = p & 63;
    float m = imask[(size_t)b * (256u*1024u) + (size_t)(h*16)*1024 + (w*16)];
    sm[p] = m;
    mask_hw[b*HW_ + p] = m;
    alpha_sum[b*HW_ + p] = 0.f;
  }
  __syncthreads();
  float s = 0.f;
  for (int p = tid; p < HW_; p += 256) s += sm[p];
  red[tid] = s; __syncthreads();
  for (int off = 128; off > 0; off >>= 1) { if (tid < off) red[tid] += red[tid+off]; __syncthreads(); }
  float inv = 1.f / red[0];
  for (int c = tid; c < C_; c += 256) {
    const float* cp = cnn + ((size_t)b * C_ + c) * HW_;
    float a = 0.f;
    for (int p = 0; p < HW_; p++) a += cp[p] * sm[p];
    avg[c] = a * inv;
  }
  __syncthreads();
  for (int j = tid; j < HID_; j += 256) {
    const float* wr = initW + (size_t)j * C_;
    float a = initb[j];
    for (int c = 0; c < C_; c++) a += avg[c] * wr[c];
    hidden[b*HID_ + j] = tanhf(a);
    const float* cw = cntW + (size_t)j * V_;
    float cc = cntb[j];
    for (int v = 0; v < V_; v++) cc += cpred[b*V_ + v] * cw[v];
    cctx[b*HID_ + j] = cc;
  }
}

// sine positional embedding (mask identical across batch -> compute once, batch 0)
__global__ void k_pos(const float* __restrict__ mask_hw, float* __restrict__ pos) {
  int tid = threadIdx.x;
  __shared__ float yn[HW_], xn[HW_];
  for (int p = tid; p < HW_; p += 256) {
    int h = p >> 6, w = p & 63;
    float y = 0.f, yt = 0.f, x = 0.f, xt = 0.f;
    for (int hh = 0; hh < H_; hh++) { float m = mask_hw[hh*W_ + w]; yt += m; if (hh <= h) y += m; }
    for (int ww = 0; ww < W_; ww++) { float m = mask_hw[h*W_ + ww]; xt += m; if (ww <= w) x += m; }
    const float TWO_PI = 6.283185307179586f;
    yn[p] = y / (yt + 1e-6f) * TWO_PI;
    xn[p] = x / (xt + 1e-6f) * TWO_PI;
  }
  __syncthreads();
  for (int idx = tid; idx < AD_*HW_; idx += 256) {
    int d = idx >> 10, p = idx & 1023;
    int i = d & 255;
    float f = powf(10000.f, (2.f * (float)(i >> 1)) / 256.f);
    float base = (d < 256 ? yn[p] : xn[p]) / f;
    pos[idx] = ((i & 1) == 0) ? sinf(base) : cosf(base);
  }
}

// cnn (B,C,HW) -> A_bf16 (M=8192, KP=704) row-major, zero-padded
__global__ void k_prepA(const float* __restrict__ cnn, __bf16* __restrict__ at) {
  int m = blockIdx.x;
  int b = m >> 10, p = m & 1023;
  const float* src = cnn + (size_t)b * C_ * HW_ + p;
  __bf16* dst = at + (size_t)m * KP_;
  for (int c = threadIdx.x; c < KP_; c += 256)
    dst[c] = (c < C_) ? (__bf16)src[(size_t)c * HW_] : (__bf16)0.f;
}

// enc_conv_w (512,684) -> B^T bf16 (N=512, KP=704)
__global__ void k_prepB(const float* __restrict__ encW, __bf16* __restrict__ bt) {
  int d = blockIdx.x;
  for (int c = threadIdx.x; c < KP_; c += 256)
    bt[(size_t)d*KP_ + c] = (c < C_) ? (__bf16)encW[(size_t)d*C_ + c] : (__bf16)0.f;
}

// fold att_weight_W into att_conv_w: K2t[d][k] = sum_c attW[d,c]*convw[c,k], (512 x 128) bf16
__global__ void k_k2(const float* __restrict__ attW, const float* __restrict__ convw,
                     __bf16* __restrict__ k2t) {
  int d = blockIdx.x, k = threadIdx.x;
  float a = 0.f;
  if (k < 121) {
    const float* wr = attW + (size_t)d * 512;
    for (int c = 0; c < 512; c++) a += wr[c] * convw[(size_t)c * 121 + k];
  }
  k2t[(size_t)d*KC_ + k] = (__bf16)a;
}

// WMMA GEMM: trans[m,n] = sum_k A[m,k]*Bt[n,k] + encb[n] + pos[n, m%1024]
__global__ void k_trans(const __bf16* __restrict__ at, const __bf16* __restrict__ bt,
                        const float* __restrict__ encb, const float* __restrict__ pos,
                        float* __restrict__ trans) {
  int lane = threadIdx.x & 31;
  int wave = threadIdx.x >> 5;
  int wg = blockIdx.x * 8 + wave;        // 16384 tiles
  int ntile = wg & 31, mtile = wg >> 5;
  int mrow = mtile * 16, nb = ntile * 16;
  v8f acc = {};
  for (int kb = 0; kb < KP_; kb += 32) {
    v16bf a  = load_a (at, KP_, mrow, kb, lane);
    v16bf bb = load_bt(bt, KP_, nb,   kb, lane);
    acc = __builtin_amdgcn_wmma_f32_16x16x32_bf16(false, a, false, bb, (short)0, acc, false, false);
  }
  int n = nb + (lane & 15), hf = lane >> 4;
  float eb = encb[n];
  for (int r = 0; r < 8; r++) {
    int m = mrow + hf * 8 + r;
    int p = m & 1023;
    trans[(size_t)m * AD_ + n] = acc[r] + eb + pos[(size_t)n * HW_ + p];
  }
}

// ---------------- per-step kernels ----------------

// Blocks 0..7: GRU cell + query + hidden/we-dependent part of out_state (per batch).
// Blocks 8..263: im2col of alpha_sum (11x11, pad 5) -> patches (M, 128) bf16.
__global__ void k_step_a(int t, const int* __restrict__ labels, const float* __restrict__ emb,
                         const float* __restrict__ wih, const float* __restrict__ whh,
                         const float* __restrict__ bih, const float* __restrict__ bhh,
                         const float* __restrict__ attW, const float* __restrict__ attb,
                         const float* __restrict__ stW, const float* __restrict__ stb,
                         const float* __restrict__ ewW, const float* __restrict__ ewb,
                         const float* __restrict__ cctx, float* __restrict__ hidden,
                         float* __restrict__ query, float* __restrict__ pout,
                         const float* __restrict__ alpha_sum, __bf16* __restrict__ patches) {
  __shared__ float we[HID_], h[HID_], gi[768], gh[768], hn[HID_];
  int tid = threadIdx.x;
  if (blockIdx.x >= B_) {
    // ---- im2col path ----
    int idx = (blockIdx.x - B_) * 256 + tid;       // 256 blocks
    for (int e = 0; e < 16; e++) {
      int id = idx + e * (256*256);
      int m = id >> 7, k = id & 127;
      int b = m >> 10, p = m & 1023, hh0 = p >> 6, ww0 = p & 63;
      float v = 0.f;
      if (k < 121) {
        int i = k / 11, j = k - i * 11;
        int hh = hh0 + i - 5, ww = ww0 + j - 5;
        if (hh >= 0 && hh < H_ && ww >= 0 && ww < W_)
          v = alpha_sum[b*HW_ + hh*W_ + ww];
      }
      patches[id] = (__bf16)v;
    }
    return;
  }
  // ---- GRU path ----
  int b = blockIdx.x;
  int word = (t == 0) ? 1 : labels[b*T_ + (t-1)];
  for (int j = tid; j < HID_; j += 256) {
    we[j] = emb[(size_t)word * HID_ + j];
    h[j]  = hidden[b*HID_ + j];
  }
  __syncthreads();
  for (int g = tid; g < 768; g += 256) {
    const float* wi = wih + (size_t)g * HID_;
    const float* wh = whh + (size_t)g * HID_;
    float si = bih[g], sh = bhh[g];
    for (int k = 0; k < HID_; k++) { si += we[k]*wi[k]; sh += h[k]*wh[k]; }
    gi[g] = si; gh[g] = sh;
  }
  __syncthreads();
  for (int j = tid; j < HID_; j += 256) {
    float r = 1.f/(1.f+expf(-(gi[j]+gh[j])));
    float z = 1.f/(1.f+expf(-(gi[256+j]+gh[256+j])));
    float n = tanhf(gi[512+j] + r*gh[512+j]);
    float v = (1.f - z)*n + z*h[j];
    hn[j] = v; hidden[b*HID_ + j] = v;
  }
  __syncthreads();
  for (int d = tid; d < AD_; d += 256) {
    const float* wr = attW + (size_t)d * HID_;
    float q = attb[d];
    for (int k = 0; k < HID_; k++) q += hn[k]*wr[k];
    query[b*AD_ + d] = q;
  }
  for (int j = tid; j < HID_; j += 256) {
    const float* ws = stW + (size_t)j * HID_;
    const float* wemb = ewW + (size_t)j * HID_;
    float o = stb[j] + ewb[j] + cctx[b*HID_ + j];
    for (int k = 0; k < HID_; k++) o += hn[k]*ws[k] + we[k]*wemb[k];
    pout[b*HID_ + j] = o;
  }
}

// Fused coverage GEMM (WMMA, K2 staged in LDS) + tanh-score + energy dot + reduce.
// 64 blocks x 256 threads; wave w handles 16 rows, all 512 columns.
__global__ void k_cov(const __bf16* __restrict__ patches, const __bf16* __restrict__ k2t,
                      const float* __restrict__ query, const float* __restrict__ trans,
                      const float* __restrict__ acW, const float* __restrict__ acb,
                      float* __restrict__ energy) {
  extern __shared__ __bf16 k2s[];                 // 512 rows x LDK2 bf16
  int tid = threadIdx.x;
  // cooperative copy K2t (512 x 128 bf16 = 8192 x 16B) into padded LDS
  for (int ch = tid; ch < 8192; ch += 256) {
    int row = ch >> 4, q = ch & 15;
    uint4 v = ((const uint4*)k2t)[ch];
    *(uint4*)((char*)k2s + (size_t)row * (LDK2*2) + q * 16) = v;
  }
  __syncthreads();

  int lane = tid & 31, wave = tid >> 5, hf = lane >> 4;
  int mrow = blockIdx.x * 128 + wave * 16;
  int b = mrow >> 10;
  v16bf af[4];
  for (int kc = 0; kc < 4; kc++) af[kc] = load_a(patches, KC_, mrow, kc*32, lane);
  float ep[8];
  for (int r = 0; r < 8; r++) ep[r] = 0.f;

  for (int nt = 0; nt < 32; nt++) {
    v8f acc = {};
    int n = nt*16 + (lane & 15);
    const __bf16* brow = &k2s[(size_t)n * LDK2];
    for (int kc = 0; kc < 4; kc++) {
      const __bf16* p = brow + kc*32 + hf*16;
      v8bf lo = *(const v8bf*)p;
      v8bf hi = *(const v8bf*)(p + 8);
      v16bf bf = cat8(lo, hi);
      acc = __builtin_amdgcn_wmma_f32_16x16x32_bf16(false, af[kc], false, bf, (short)0, acc, false, false);
    }
    float q  = query[b*AD_ + n];
    float aw = acW[n];
    const float* tr = trans + (size_t)(mrow + hf*8) * AD_ + n;
    for (int r = 0; r < 8; r++) {
      float sc = tanhf(q + acc[r] + tr[(size_t)r * AD_]);
      ep[r] += sc * aw;
    }
  }
  // reduce across the 16 lanes sharing a row group
  for (int msk = 1; msk < 16; msk <<= 1)
    for (int r = 0; r < 8; r++) ep[r] += __shfl_xor(ep[r], msk, 32);
  if ((lane & 15) == 0) {
    float bias = acb[0];
    for (int r = 0; r < 8; r++) energy[mrow + hf*8 + r] = ep[r] + bias;
  }
}

// softmax (with redundant per-block global max), alpha_sum update, ctx, out_state, prob.
// One block per batch; each block scans all 8192 energies for the global max
// (cheaper than an extra reduction-kernel launch on the serial chain).
__global__ void k_actx(int t, const float* __restrict__ energy, const float* __restrict__ mask_hw,
                       const float* __restrict__ cnn, float* __restrict__ alpha_sum,
                       const float* __restrict__ ctxW, const float* __restrict__ ctxb,
                       const float* __restrict__ pout, const float* __restrict__ outW,
                       const float* __restrict__ outb, float* __restrict__ dout) {
  int b = blockIdx.x, tid = threadIdx.x;
  __shared__ float alpha[HW_];
  __shared__ float ctx[C_];
  __shared__ float os[HID_];
  __shared__ float red[256];
  // global max over all batches
  float m = -3.4e38f;
  for (int i = tid; i < M_; i += 256) m = fmaxf(m, energy[i]);
  red[tid] = m; __syncthreads();
  for (int off = 128; off > 0; off >>= 1) { if (tid < off) red[tid] = fmaxf(red[tid], red[tid+off]); __syncthreads(); }
  float gmax = red[0];
  __syncthreads();
  // this batch's masked exp-sum
  float s = 0.f;
  for (int p = tid; p < HW_; p += 256) {
    int i = b*HW_ + p;
    float e = expf(energy[i] - gmax) * mask_hw[i];
    alpha[p] = e;
    s += e;
  }
  red[tid] = s; __syncthreads();
  for (int off = 128; off > 0; off >>= 1) { if (tid < off) red[tid] += red[tid+off]; __syncthreads(); }
  float inv = 1.f / (red[0] + 1e-10f);
  for (int p = tid; p < HW_; p += 256) {
    float a = alpha[p] * inv;
    alpha[p] = a;
    alpha_sum[b*HW_ + p] += a;
  }
  __syncthreads();
  for (int c = tid; c < C_; c += 256) {
    const float* cp = cnn + ((size_t)b*C_ + c) * HW_;
    float sc = 0.f;
    for (int p = 0; p < HW_; p++) sc += alpha[p] * cp[p];
    ctx[c] = sc;
  }
  __syncthreads();
  for (int j = tid; j < HID_; j += 256) {
    const float* wr = ctxW + (size_t)j * C_;
    float o = pout[b*HID_ + j] + ctxb[j];
    for (int c = 0; c < C_; c++) o += ctx[c]*wr[c];
    os[j] = o;
  }
  __syncthreads();
  for (int v = tid; v < V_; v += 256) {
    const float* wr = outW + (size_t)v * HID_;
    float p = outb[v];
    for (int k = 0; k < HID_; k++) p += os[k]*wr[k];
    dout[((size_t)b*T_ + t)*V_ + v] = p;
  }
}

extern "C" void kernel_launch(void* const* d_in, const int* in_sizes, int n_in,
                              void* d_out, int out_size, void* d_ws, size_t ws_size,
                              hipStream_t stream) {
  (void)in_sizes; (void)n_in; (void)out_size; (void)ws_size;
  const float* cnn   = (const float*)d_in[0];
  const float* cpred = (const float*)d_in[1];
  const float* imask = (const float*)d_in[2];
  const float* initW = (const float*)d_in[3];
  const float* initb = (const float*)d_in[4];
  const float* emb   = (const float*)d_in[5];
  const float* wih   = (const float*)d_in[6];
  const float* whh   = (const float*)d_in[7];
  const float* bih   = (const float*)d_in[8];
  const float* bhh   = (const float*)d_in[9];
  const float* attW  = (const float*)d_in[10];
  const float* attb  = (const float*)d_in[11];
  const float* convw = (const float*)d_in[12];
  const float* awW   = (const float*)d_in[13];
  const float* acW   = (const float*)d_in[14];
  const float* acb   = (const float*)d_in[15];
  const float* encW  = (const float*)d_in[16];
  const float* encb  = (const float*)d_in[17];
  const float* stW   = (const float*)d_in[18];
  const float* stb   = (const float*)d_in[19];
  const float* ewW   = (const float*)d_in[20];
  const float* ewb   = (const float*)d_in[21];
  const float* ctxW  = (const float*)d_in[22];
  const float* ctxb  = (const float*)d_in[23];
  const float* cntW  = (const float*)d_in[24];
  const float* cntb  = (const float*)d_in[25];
  const float* outW  = (const float*)d_in[26];
  const float* outb  = (const float*)d_in[27];
  const int*   labels= (const int*)d_in[28];
  float* dout = (float*)d_out;

  char* ws = (char*)d_ws;
  size_t off = 0;
  auto alloc = [&](size_t bytes) -> void* {
    void* p = ws + off;
    off = (off + bytes + 255) & ~(size_t)255;
    return p;
  };
  float*  pos      = (float*) alloc((size_t)AD_*HW_*4);     // 2 MB
  float*  trans    = (float*) alloc((size_t)M_*AD_*4);      // 16 MB
  __bf16* at       = (__bf16*)alloc((size_t)M_*KP_*2);      // 11.5 MB
  __bf16* bt       = (__bf16*)alloc((size_t)AD_*KP_*2);     // 0.7 MB
  __bf16* k2t      = (__bf16*)alloc((size_t)AD_*KC_*2);     // 128 KB
  __bf16* patches  = (__bf16*)alloc((size_t)M_*KC_*2);      // 2 MB
  float*  mask_hw  = (float*) alloc((size_t)B_*HW_*4);
  float*  hidden   = (float*) alloc((size_t)B_*HID_*4);
  float*  cctx     = (float*) alloc((size_t)B_*HID_*4);
  float*  alpha_sum= (float*) alloc((size_t)B_*HW_*4);
  float*  query    = (float*) alloc((size_t)B_*AD_*4);
  float*  pout     = (float*) alloc((size_t)B_*HID_*4);
  float*  energy   = (float*) alloc((size_t)M_*4);

  // one-time precompute
  k_init <<<B_,  256, 0, stream>>>(cnn, cpred, imask, initW, initb, cntW, cntb,
                                   mask_hw, hidden, cctx, alpha_sum);
  k_pos  <<<1,   256, 0, stream>>>(mask_hw, pos);
  k_prepA<<<M_,  256, 0, stream>>>(cnn, at);
  k_prepB<<<AD_, 256, 0, stream>>>(encW, bt);
  k_k2   <<<AD_, KC_, 0, stream>>>(awW, convw, k2t);
  k_trans<<<2048,256, 0, stream>>>(at, bt, encb, pos, trans);

  for (int t = 0; t < T_; t++) {
    k_step_a<<<B_ + 256, 256, 0, stream>>>(t, labels, emb, wih, whh, bih, bhh,
                                           attW, attb, stW, stb, ewW, ewb,
                                           cctx, hidden, query, pout,
                                           alpha_sum, patches);
    k_cov   <<<64,  256, (size_t)AD_*LDK2*2, stream>>>(patches, k2t, query, trans,
                                                       acW, acb, energy);
    k_actx  <<<B_,  256, 0, stream>>>(t, energy, mask_hw, cnn, alpha_sum,
                                      ctxW, ctxb, pout, outW, outb, dout);
  }
}
